// RRTAttention_34428457844867
// MI455X (gfx1250) — matrix operand order repaired and checked by reading, not compiled, not run on hardware
//
#include <hip/hip_runtime.h>
#include <hip/hip_bf16.h>

// ---------- problem constants ----------
#define DIMX    1024
#define NHEADS  16
#define NKV     4
#define HD      64
#define KVDIM   256
#define SEQ     2048
#define BATCH   2
#define MTOT    (BATCH*SEQ)   // 4096
#define LORA_R  4
#define LORA_STEPS 2

// ---------- CDNA5 vector types ----------
typedef __attribute__((ext_vector_type(16))) __bf16          v16bf;
typedef __attribute__((ext_vector_type(8)))  float           v8f;
typedef __attribute__((ext_vector_type(8)))  unsigned short  u16x8;
typedef __attribute__((ext_vector_type(16))) unsigned short  u16x16;
typedef __attribute__((ext_vector_type(4)))  unsigned int    u32x4;
typedef __attribute__((ext_vector_type(4)))  int             i32x4;
typedef __attribute__((ext_vector_type(8)))  int             i32x8;

#if __has_builtin(__builtin_amdgcn_tensor_load_to_lds) && __has_builtin(__builtin_amdgcn_s_wait_tensorcnt)
#define HAVE_TDM 1
#else
#define HAVE_TDM 0
#endif

__device__ __forceinline__ unsigned short f2bf(float f) {
  __bf16 b = (__bf16)f;                      // native cvt (RNE)
  return __builtin_bit_cast(unsigned short, b);
}
__device__ __forceinline__ v16bf cat_bf(u16x8 lo, u16x8 hi) {
  u16x16 u = __builtin_shufflevector(lo, hi, 0,1,2,3,4,5,6,7,8,9,10,11,12,13,14,15);
  return __builtin_bit_cast(v16bf, u);
}
__device__ __forceinline__ v16bf as_bf(u16x16 u) { return __builtin_bit_cast(v16bf, u); }

__device__ __forceinline__ v8f wmma_bf16(v16bf a, v16bf b, v8f c) {
  // (neg_a, A, neg_b, B, c_mod, C, reuse_a, reuse_b)
  return __builtin_amdgcn_wmma_f32_16x16x32_bf16(false, a, false, b, (short)0, c,
                                                 false, false);
}

#if HAVE_TDM
// Issue one TDM 2D tile load: tile_d1 rows of tile_d0 bf16 elements,
// global row stride = stride_elems, packed contiguously into LDS at lds_off.
__device__ __forceinline__ void tdm_load_2d(unsigned lds_off, const void* gp,
                                            unsigned tile_d0, unsigned tile_d1,
                                            unsigned long long stride_elems) {
  unsigned long long ga = (unsigned long long)(size_t)gp;
  u32x4 g0;
  g0[0] = 1u;                                          // count=1 (valid user D#)
  g0[1] = lds_off;                                     // lds_addr (bytes)
  g0[2] = (unsigned)(ga & 0xFFFFFFFFu);                // global_addr[31:0]
  g0[3] = (unsigned)((ga >> 32) & 0x01FFFFFFu) | (2u << 30);  // addr[56:32], type=2
  i32x8 g1;
  g1[0] = (int)(1u << 16);                             // wg_mask=0, data_size=2B
  g1[1] = (int)((tile_d0 & 0xFFFFu) << 16);            // tensor_dim0[15:0]
  g1[2] = (int)(((tile_d0 >> 16) & 0xFFFFu) | ((tile_d1 & 0xFFFFu) << 16));
  g1[3] = (int)(((tile_d1 >> 16) & 0xFFFFu) | ((tile_d0 & 0xFFFFu) << 16)); // tile_dim0
  g1[4] = (int)(tile_d1 & 0xFFFFu);                    // tile_dim1, tile_dim2=0
  g1[5] = (int)(stride_elems & 0xFFFFFFFFull);         // tensor_dim0_stride[31:0]
  g1[6] = (int)((stride_elems >> 32) & 0xFFFFull);     // stride[47:32], dim1_stride lo=0
  g1[7] = 0;
  i32x4 z4 = {0, 0, 0, 0};
#if defined(__clang_major__) && (__clang_major__ >= 23)
  i32x8 z8 = {0, 0, 0, 0, 0, 0, 0, 0};
  __builtin_amdgcn_tensor_load_to_lds(g0, g1, z4, z4, z8, 0);
#else
  __builtin_amdgcn_tensor_load_to_lds(g0, g1, z4, z4, 0);
#endif
}
#endif

// =====================================================================
// elementwise f32 -> bf16 (n multiple of 8)
// =====================================================================
__global__ __launch_bounds__(256)
void f32_to_bf16_kernel(const float* __restrict__ src, unsigned short* __restrict__ dst,
                        long n)
{
  long i = ((long)blockIdx.x * 256 + threadIdx.x) * 8;
  if (i >= n) return;
  float4 a = *(const float4*)(src + i);
  float4 b = *(const float4*)(src + i + 4);
  u16x8 v;
  v[0] = f2bf(a.x); v[1] = f2bf(a.y); v[2] = f2bf(a.z); v[3] = f2bf(a.w);
  v[4] = f2bf(b.x); v[5] = f2bf(b.y); v[6] = f2bf(b.z); v[7] = f2bf(b.w);
  *(u16x8*)(dst + i) = v;
}

// =====================================================================
// GEMM: C[M,N] f32 = Xbf[M,K] * Wbf[N,K]^T   (bf16 in, WMMA, TDM staging)
// 256 thr = 8 waves (4xM, 2xN), tile 128x128, K step 32, double-buffered LDS.
// =====================================================================
__global__ __launch_bounds__(256)
void gemm_bf16_kernel(const unsigned short* __restrict__ X,
                      const unsigned short* __restrict__ W,
                      float* __restrict__ C, int M, int N, int K)
{
  __shared__ unsigned short Xs[2][128][32];
  __shared__ unsigned short Ws[2][128][32];
  const int tid  = threadIdx.x;
  const int lane = tid & 31, wid = tid >> 5;
  const int hi   = lane >> 4, l15 = lane & 15;
  const int waveM = wid & 3, waveN = wid >> 2;
  const int bm = blockIdx.y * 128, bn = blockIdx.x * 128;
  const int niter = K >> 5;

  v8f acc[2][4];
#pragma unroll
  for (int i = 0; i < 2; i++)
#pragma unroll
    for (int j = 0; j < 4; j++)
#pragma unroll
      for (int r = 0; r < 8; r++) acc[i][j][r] = 0.f;

  auto stage = [&](int i, int buf) {
#if HAVE_TDM
    if (wid == 0) {
      tdm_load_2d((unsigned)(size_t)(void*)&Xs[buf][0][0],
                  X + (size_t)bm * K + i * 32, 32u, 128u, (unsigned long long)K);
      tdm_load_2d((unsigned)(size_t)(void*)&Ws[buf][0][0],
                  W + (size_t)bn * K + i * 32, 32u, 128u, (unsigned long long)K);
    }
#else
#pragma unroll
    for (int j = 0; j < 2; j++) {
      int e = tid * 16 + j * 8;
      int r = e >> 5, c = e & 31;
      *(u16x8*)&Xs[buf][r][c] = *(const u16x8*)(X + (size_t)(bm + r) * K + i * 32 + c);
      *(u16x8*)&Ws[buf][r][c] = *(const u16x8*)(W + (size_t)(bn + r) * K + i * 32 + c);
    }
#endif
  };

  stage(0, 0);
  for (int i = 0; i < niter; i++) {
#if HAVE_TDM
    if (wid == 0) __builtin_amdgcn_s_wait_tensorcnt(0);
#endif
    __syncthreads();                 // tile i visible; buf[(i+1)&1] free
    if (i + 1 < niter) stage(i + 1, (i + 1) & 1);
    const int buf = i & 1;

    v16bf a[2];
#pragma unroll
    for (int mi = 0; mi < 2; mi++) {
      int m  = waveM * 32 + mi * 16 + l15;
      int kk = hi * 8;
      u16x8 lo = *(const u16x8*)&Xs[buf][m][kk];
      u16x8 h8 = *(const u16x8*)&Xs[buf][m][kk + 16];
      a[mi] = cat_bf(lo, h8);
    }
#pragma unroll
    for (int ni = 0; ni < 4; ni++) {
      int n = waveN * 64 + ni * 16 + l15;
      v16bf b = cat_bf(*(const u16x8*)&Ws[buf][n][hi * 8], *(const u16x8*)&Ws[buf][n][hi * 8 + 16]);
#pragma unroll
      for (int mi = 0; mi < 2; mi++)
        acc[mi][ni] = wmma_bf16(a[mi], b, acc[mi][ni]);
    }
  }

#pragma unroll
  for (int mi = 0; mi < 2; mi++)
#pragma unroll
    for (int ni = 0; ni < 4; ni++)
#pragma unroll
      for (int r = 0; r < 8; r++) {
        int row = bm + waveM * 32 + mi * 16 + r + hi * 8;
        int col = bn + waveN * 64 + ni * 16 + l15;
        C[(size_t)row * N + col] = acc[mi][ni][r];
      }
}

// =====================================================================
// LoRA: T[M,4] = X @ A[step]^T ; then C += T @ B[step]^T (scale=1)
// =====================================================================
__global__ __launch_bounds__(256)
void lora_t_kernel(const float* __restrict__ X, const float* __restrict__ A,
                   const int* __restrict__ step, float* __restrict__ T, int M, int K)
{
  int idx = blockIdx.x * 256 + threadIdx.x;
  if (idx >= M * LORA_R) return;
  int m = idx >> 2, r = idx & 3;
  int st = *step;
  float acc = 0.f;
  if (st >= 0 && st < LORA_STEPS) {
    const float* a  = A + (size_t)(st * LORA_R + r) * K;
    const float* xr = X + (size_t)m * K;
    for (int k = 0; k < K; k++) acc += xr[k] * a[k];
  }
  T[idx] = acc;
}

__global__ __launch_bounds__(256)
void lora_add_kernel(float* __restrict__ C, const float* __restrict__ T,
                     const float* __restrict__ Bw, const int* __restrict__ step,
                     int M, int N)
{
  int idx = blockIdx.x * 256 + threadIdx.x;
  if (idx >= M * N) return;
  int st = *step;
  if (st < 0 || st >= LORA_STEPS) return;
  int m = idx / N, n = idx - m * N;
  const float* t = T + (size_t)m * 4;
  const float* b = Bw + (size_t)st * N * 4 + (size_t)n * 4;
  C[idx] += t[0] * b[0] + t[1] * b[1] + t[2] * b[2] + t[3] * b[3];
}

// =====================================================================
// RMSNorm + partial RoPE (+gain) + bf16 relayout [B,S,NH,64] -> [B,NH,S,64]
// =====================================================================
__global__ __launch_bounds__(256)
void norm_rope_bf16_kernel(const float* __restrict__ src, unsigned short* __restrict__ dst,
                           const float* __restrict__ gain, int NH, int do_norm_rope)
{
  const int lane = threadIdx.x & 31;
  const int wid  = threadIdx.x >> 5;
  const long row = (long)blockIdx.x * 8 + wid;
  const int h = (int)(row % NH);
  const long t = row / NH;
  const int s = (int)(t % SEQ);
  const int b = (int)(t / SEQ);
  const float* xr = src + row * HD;
  float x0 = xr[lane], x1 = xr[lane + 32];
  if (do_norm_rope) {
    float ss = x0 * x0 + x1 * x1;
#pragma unroll
    for (int m = 1; m < 32; m <<= 1) ss += __shfl_xor(ss, m, 32);
    float rn = rsqrtf(ss * (1.f / 64.f) + 1.1920929e-7f);
    x0 *= rn; x1 *= rn;
    float xo = __shfl_xor(x0, 16, 32);
    int   i  = lane & 15;
    float invf = __expf(-(float)i * 0.5756462732485115f);  // ln(10000)/16
    float fr = (float)s * invf;
    float sn, cs;
    __sincosf(fr, &sn, &cs);
    x0 = (lane < 16) ? (x0 * cs + xo * sn) : (x0 * cs - xo * sn);
  }
  if (gain) { float g = gain[h]; x0 *= g; x1 *= g; }
  unsigned short* dp = dst + (((long)b * NH + h) * SEQ + s) * HD;
  dp[lane]      = f2bf(x0);
  dp[lane + 32] = f2bf(x1);
}

// =====================================================================
// Causal flash attention, bf16 WMMA, f32 online softmax, bf16 output.
// 256 thr = 8 waves, 128 q rows / block, key tile 64 staged via TDM.
// =====================================================================
__global__ __launch_bounds__(256)
void flash_attn_kernel(const unsigned short* __restrict__ Q,
                       const unsigned short* __restrict__ Kb,
                       const unsigned short* __restrict__ Vb,
                       unsigned short* __restrict__ Yb)
{
  __shared__ unsigned short Ks[64][64];        // [key][d]
  __shared__ unsigned short Vt[64][64];        // [d][key] transposed at load
  __shared__ unsigned short Ps[8][16][64];     // per-wave P tile bounce
  const int tid  = threadIdx.x;
  const int lane = tid & 31, wid = tid >> 5;
  const int hi   = lane >> 4, l15 = lane & 15;
  const int b = blockIdx.y >> 4, h = blockIdx.y & 15, hk = h >> 2;
  const int qBase = blockIdx.x * 128;
  const int qRow0 = qBase + wid * 16;

  v16bf aq[2];
  {
    const unsigned short* qp = Q + (((size_t)b * NHEADS + h) * SEQ + qRow0 + l15) * HD;
#pragma unroll
    for (int c = 0; c < 2; c++) {
      int d0 = c * 32 + hi * 8;
      u16x8 lo = *(const u16x8*)(qp + d0);
      u16x8 h8 = *(const u16x8*)(qp + d0 + 16);
      aq[c] = cat_bf(lo, h8);
    }
  }

  v8f accO[4];
#pragma unroll
  for (int j = 0; j < 4; j++)
#pragma unroll
    for (int r = 0; r < 8; r++) accO[j][r] = 0.f;
  float rowm[8], rows[8];
#pragma unroll
  for (int r = 0; r < 8; r++) { rowm[r] = -1e30f; rows[r] = 0.f; }

  const unsigned short* kbase = Kb + ((size_t)b * NKV + hk) * SEQ * HD;
  const unsigned short* vbase = Vb + ((size_t)b * NKV + hk) * SEQ * HD;
  const int nKB = (qBase >> 6) + 2;

  for (int jb = 0; jb < nKB; ++jb) {
    const int kBase = jb * 64;
    // stage K tile (TDM: one DMA of 64x64 contiguous bf16)
#if HAVE_TDM
    if (wid == 0)
      tdm_load_2d((unsigned)(size_t)(void*)&Ks[0][0],
                  kbase + (size_t)kBase * HD, 64u, 64u, 64ull);
#else
#pragma unroll
    for (int j = 0; j < 2; j++) {
      int e = tid * 16 + j * 8;
      int kk = e >> 6, d = e & 63;
      *(u16x8*)&Ks[kk][d] = *(const u16x8*)(kbase + (size_t)(kBase + kk) * HD + d);
    }
#endif
    // stage V tile transposed (manual scatter)
#pragma unroll
    for (int j = 0; j < 2; j++) {
      int e = tid * 16 + j * 8;
      int kk = e >> 6, d = e & 63;
      u16x8 vv = *(const u16x8*)(vbase + (size_t)(kBase + kk) * HD + d);
#pragma unroll
      for (int i = 0; i < 8; i++) Vt[d + i][kk] = vv[i];
    }
#if HAVE_TDM
    if (wid == 0) __builtin_amdgcn_s_wait_tensorcnt(0);
#endif
    __syncthreads();

    // S = Q K^T * scale, causal mask
    float p[4][8];
#pragma unroll
    for (int ni = 0; ni < 4; ni++) {
      v8f accS;
#pragma unroll
      for (int r = 0; r < 8; r++) accS[r] = 0.f;
#pragma unroll
      for (int c = 0; c < 2; c++) {
        v16bf bK = cat_bf(*(const u16x8*)&Ks[ni * 16 + l15][c * 32 + hi * 8], *(const u16x8*)&Ks[ni * 16 + l15][c * 32 + hi * 8 + 16]);
        accS = wmma_bf16(aq[c], bK, accS);
      }
      int kcol = kBase + ni * 16 + l15;
#pragma unroll
      for (int r = 0; r < 8; r++) {
        int qi = qRow0 + r + hi * 8;
        float sv = accS[r] * 0.125f;          // 1/sqrt(64)
        p[ni][r] = (kcol <= qi) ? sv : -1e30f;
      }
    }

    // online softmax per row
#pragma unroll
    for (int r = 0; r < 8; r++) {
      float mr = fmaxf(fmaxf(p[0][r], p[1][r]), fmaxf(p[2][r], p[3][r]));
      mr = fmaxf(mr, __shfl_xor(mr, 1, 32));
      mr = fmaxf(mr, __shfl_xor(mr, 2, 32));
      mr = fmaxf(mr, __shfl_xor(mr, 4, 32));
      mr = fmaxf(mr, __shfl_xor(mr, 8, 32));
      float mnew = fmaxf(rowm[r], mr);
      float corr = __expf(rowm[r] - mnew);
      rowm[r] = mnew;
      float rsum = 0.f;
#pragma unroll
      for (int ni = 0; ni < 4; ni++) {
        float e = __expf(p[ni][r] - mnew);
        rsum += e;
        Ps[wid][r + hi * 8][ni * 16 + l15] = f2bf(e);
      }
      rsum += __shfl_xor(rsum, 1, 32);
      rsum += __shfl_xor(rsum, 2, 32);
      rsum += __shfl_xor(rsum, 4, 32);
      rsum += __shfl_xor(rsum, 8, 32);
      rows[r] = rows[r] * corr + rsum;
#pragma unroll
      for (int nj = 0; nj < 4; nj++) accO[nj][r] *= corr;
    }

    // O += P @ V
#pragma unroll
    for (int c = 0; c < 2; c++) {
      u16x8 lo = *(const u16x8*)&Ps[wid][l15][c * 32 + hi * 8];
      u16x8 h8 = *(const u16x8*)&Ps[wid][l15][c * 32 + hi * 8 + 16];
      v16bf ap = cat_bf(lo, h8);
#pragma unroll
      for (int nj = 0; nj < 4; nj++) {
        v16bf bV = cat_bf(*(const u16x8*)&Vt[nj * 16 + l15][c * 32 + hi * 8], *(const u16x8*)&Vt[nj * 16 + l15][c * 32 + hi * 8 + 16]);
        accO[nj] = wmma_bf16(ap, bV, accO[nj]);
      }
    }
    __syncthreads();
  }

  // write y[b, s, h*64 + d] as bf16 (feeds out-proj GEMM directly)
#pragma unroll
  for (int r = 0; r < 8; r++) {
    float inv = 1.0f / rows[r];
    int qi = qRow0 + r + hi * 8;
    unsigned short* yp = Yb + ((size_t)b * SEQ + qi) * DIMX + h * HD;
#pragma unroll
    for (int nj = 0; nj < 4; nj++)
      yp[nj * 16 + l15] = f2bf(accO[nj][r] * inv);
  }
}

// =====================================================================
// launcher
// =====================================================================
extern "C" void kernel_launch(void* const* d_in, const int* in_sizes, int n_in,
                              void* d_out, int out_size, void* d_ws, size_t ws_size,
                              hipStream_t stream) {
  (void)in_sizes; (void)n_in; (void)out_size; (void)ws_size;
  const float* x     = (const float*)d_in[0];
  const float* Wq    = (const float*)d_in[1];
  const float* Wk    = (const float*)d_in[2];
  const float* Wv    = (const float*)d_in[3];
  const float* Wproj = (const float*)d_in[4];
  const float* qgain = (const float*)d_in[5];
  const float* Aq    = (const float*)d_in[6];
  const float* Bq    = (const float*)d_in[7];
  const float* Av    = (const float*)d_in[8];
  const float* Bv    = (const float*)d_in[9];
  const int*   step  = (const int*)d_in[10];
  float* out = (float*)d_out;

  char* ws = (char*)d_ws;
  float*          qf   = (float*)(ws + 0);                  // 16 MB
  float*          kf   = (float*)(ws + 16777216);           //  4 MB
  float*          vf   = (float*)(ws + 20971520);           //  4 MB
  float*          tq   = (float*)(ws + 25165824);           // 64 KB
  float*          tv   = (float*)(ws + 25231360);           // 64 KB
  unsigned short* qh   = (unsigned short*)(ws + 25296896);  //  8 MB
  unsigned short* khh  = (unsigned short*)(ws + 33685504);  //  2 MB
  unsigned short* vhh  = (unsigned short*)(ws + 35782656);  //  2 MB
  unsigned short* xbf  = (unsigned short*)(ws + 37879808);  //  8 MB
  unsigned short* wqbf = (unsigned short*)(ws + 46268416);  //  2 MB
  unsigned short* wkbf = (unsigned short*)(ws + 48365568);  // 512 KB
  unsigned short* wvbf = (unsigned short*)(ws + 48889856);  // 512 KB
  unsigned short* wpbf = (unsigned short*)(ws + 49414144);  //  2 MB
  unsigned short* ybf  = xbf;  // xbf dead after QKV GEMMs; reuse for attn output

  dim3 blk(256);
  // one-time bf16 conversions
  f32_to_bf16_kernel<<<(MTOT * DIMX / 8) / 256, blk, 0, stream>>>(x, xbf, (long)MTOT * DIMX);
  f32_to_bf16_kernel<<<(DIMX * DIMX / 8) / 256, blk, 0, stream>>>(Wq, wqbf, (long)DIMX * DIMX);
  f32_to_bf16_kernel<<<(KVDIM * DIMX / 8) / 256, blk, 0, stream>>>(Wk, wkbf, (long)KVDIM * DIMX);
  f32_to_bf16_kernel<<<(KVDIM * DIMX / 8) / 256, blk, 0, stream>>>(Wv, wvbf, (long)KVDIM * DIMX);
  f32_to_bf16_kernel<<<(DIMX * DIMX / 8) / 256, blk, 0, stream>>>(Wproj, wpbf, (long)DIMX * DIMX);
  // QKV projections (bf16 WMMA, TDM-staged tiles)
  gemm_bf16_kernel<<<dim3(DIMX / 128, MTOT / 128), blk, 0, stream>>>(xbf, wqbf, qf, MTOT, DIMX, DIMX);
  gemm_bf16_kernel<<<dim3(KVDIM / 128, MTOT / 128), blk, 0, stream>>>(xbf, wkbf, kf, MTOT, KVDIM, DIMX);
  gemm_bf16_kernel<<<dim3(KVDIM / 128, MTOT / 128), blk, 0, stream>>>(xbf, wvbf, vf, MTOT, KVDIM, DIMX);
  // LoRA on q and v (f32, rank 4)
  lora_t_kernel<<<(MTOT * LORA_R + 255) / 256, blk, 0, stream>>>(x, Aq, step, tq, MTOT, DIMX);
  lora_add_kernel<<<(MTOT * DIMX) / 256, blk, 0, stream>>>(qf, tq, Bq, step, MTOT, DIMX);
  lora_t_kernel<<<(MTOT * LORA_R + 255) / 256, blk, 0, stream>>>(x, Av, step, tv, MTOT, DIMX);
  lora_add_kernel<<<(MTOT * KVDIM) / 256, blk, 0, stream>>>(vf, tv, Bv, step, MTOT, KVDIM);
  // RMSNorm + RoPE (+gain) + bf16 head relayout
  norm_rope_bf16_kernel<<<(MTOT * NHEADS) / 8, blk, 0, stream>>>(qf, qh, qgain, NHEADS, 1);
  norm_rope_bf16_kernel<<<(MTOT * NKV) / 8, blk, 0, stream>>>(kf, khh, nullptr, NKV, 1);
  norm_rope_bf16_kernel<<<(MTOT * NKV) / 8, blk, 0, stream>>>(vf, vhh, nullptr, NKV, 0);
  // causal flash attention (writes bf16 y)
  flash_attn_kernel<<<dim3(SEQ / 128, BATCH * NHEADS), blk, 0, stream>>>(qh, khh, vhh, ybf);
  // output projection
  gemm_bf16_kernel<<<dim3(DIMX / 128, MTOT / 128), blk, 0, stream>>>(ybf, wpbf, out, MTOT, DIMX, DIMX);
}
